// TransformerBlock_14585708937480
// MI455X (gfx1250) — compile-verified, matrix operand order checked by reading, not run on hardware
//
#include <hip/hip_runtime.h>
#include <hip/hip_bf16.h>
#include <math.h>

// ---------------------------------------------------------------------------
// Transformer block for MI455X (gfx1250, wave32).
//  - All matmuls (QKV, QK^T, PV, proj, FC1, FC2) via v_wmma_f32_16x16x32_bf16.
//  - GEMM operand staging via global_load_async_to_lds_b128 (ASYNCcnt path)
//    with double-buffered LDS: tile k+1 loads overlap tile k WMMAs.
//  - LN / softmax / GELU / residuals in fp32 on the VALU.
// ---------------------------------------------------------------------------

typedef __bf16 bf16;
typedef __attribute__((ext_vector_type(16))) __bf16 v16bf;
typedef __attribute__((ext_vector_type(8)))  __bf16 v8bf;
typedef __attribute__((ext_vector_type(4)))  __bf16 v4bf;
typedef __attribute__((ext_vector_type(8)))  float  v8f;

#define Bdim 4
#define Nseq 2048
#define Cdim 768
#define Hn   12
#define HD   64
#define DFF  3072
#define NTOK (Bdim * Nseq)   // 8192 tokens

// ---- CDNA5 async copy helpers (ASYNCcnt-tracked global -> LDS DMA) --------
// GVS mode: addr = SGPR64 base + VGPR32 byte offset + inst offset.
// dst operand is a VGPR holding the LDS byte address.
__device__ __forceinline__ void async_copy64(unsigned lds, unsigned goff,
                                             const void* base) {
  asm volatile(
      "global_load_async_to_lds_b128 %0, %1, %2 offset:0\n\t"
      "global_load_async_to_lds_b128 %0, %1, %2 offset:16\n\t"
      "global_load_async_to_lds_b128 %0, %1, %2 offset:32\n\t"
      "global_load_async_to_lds_b128 %0, %1, %2 offset:48"
      :: "v"(lds), "v"(goff), "s"(base) : "memory");
}
__device__ __forceinline__ void async_copy32(unsigned lds, unsigned goff,
                                             const void* base) {
  asm volatile(
      "global_load_async_to_lds_b128 %0, %1, %2 offset:0\n\t"
      "global_load_async_to_lds_b128 %0, %1, %2 offset:16"
      :: "v"(lds), "v"(goff), "s"(base) : "memory");
}
__device__ __forceinline__ void wait_async0() {
  asm volatile("s_wait_asynccnt 0" ::: "memory");
}
__device__ __forceinline__ unsigned lds_addr_of(const void* p) {
  // Generic LDS pointers keep the wave-relative LDS byte address in [31:0].
  return (unsigned)(uintptr_t)p;
}

// ---- WMMA helpers ---------------------------------------------------------
// 16x32 bf16 A/B fragment from LDS (row-major, contraction contiguous):
// lanes 0-15 hold k=0..7 / 16..23, lanes 16-31 hold k=8..15 / 24..31 ->
// two contiguous 16B chunks per lane at byte offsets {hf*16, 32+hf*16}.
__device__ __forceinline__ v16bf load_frag(const bf16* rb, int hf) {
  v8bf lo = *(const v8bf*)(rb + hf * 8);
  v8bf hi = *(const v8bf*)(rb + 16 + hf * 8);
  return __builtin_shufflevector(lo, hi, 0,1,2,3,4,5,6,7,8,9,10,11,12,13,14,15);
}

__device__ __forceinline__ v8f wmma_bf16(v16bf a, v16bf b, v8f c) {
  return __builtin_amdgcn_wmma_f32_16x16x32_bf16(false, a, false, b, (short)0, c,
                                                 false, false);
}

// ---- elementwise fp32 -> bf16 ---------------------------------------------
__global__ __launch_bounds__(256) void cvt_k(const float* __restrict__ in,
                                             bf16* __restrict__ out, int n) {
  int i = (blockIdx.x * 256 + threadIdx.x) * 4;
  if (i < n) {
    float4 f = *(const float4*)(in + i);
    v4bf o;
    o.x = (bf16)f.x; o.y = (bf16)f.y; o.z = (bf16)f.z; o.w = (bf16)f.w;
    *(v4bf*)(out + i) = o;
  }
}

// ---- W[K,N] fp32 -> Wt[N,K] bf16 (32x32 LDS tile transpose) ---------------
__global__ __launch_bounds__(256) void transpose_cvt_k(const float* __restrict__ W,
                                                       bf16* __restrict__ Wt,
                                                       int K, int N) {
  __shared__ float tile[32][33];
  const int n0 = blockIdx.x * 32, k0 = blockIdx.y * 32;
  const int tx = threadIdx.x & 31, ty = threadIdx.x >> 5;
  for (int i = ty; i < 32; i += 8)
    tile[i][tx] = W[(size_t)(k0 + i) * N + n0 + tx];
  __syncthreads();
  for (int i = ty; i < 32; i += 8)
    Wt[(size_t)(n0 + i) * K + k0 + tx] = (bf16)tile[tx][i];
}

// ---- generic WMMA GEMM: out = epi(A[M,K] * Wt[N,K]^T + bias) --------------
// 128x128 tile / block, K-step 64, double-buffered async LDS staging.
#define GLDS2 72  // 64 + 8 pad -> 144B rows (16B aligned)

__global__ __launch_bounds__(256) void gemm_bf16_k(
    const bf16* __restrict__ A, const bf16* __restrict__ Bt,
    const float* __restrict__ bias, const float* __restrict__ resid,
    float* __restrict__ outF, bf16* __restrict__ outB,
    int M, int K, int N, int do_gelu) {
  __shared__ bf16 as[2][128 * GLDS2];
  __shared__ bf16 bs[2][128 * GLDS2];
  const int tid  = threadIdx.x;
  const int lane = tid & 31, wv = tid >> 5;
  const int wr = (wv & 3) * 32;   // wave's row offset inside 128x128 tile
  const int wc = (wv >> 2) * 64;  // wave's col offset
  const int m0 = blockIdx.y * 128, n0 = blockIdx.x * 128;
  const int fr = lane & 15, hf = lane >> 4;
  const int ldr = tid >> 1, ldo = (tid & 1) * 32;  // loader: row, elem (64B)

  const unsigned aoff = ((unsigned)(m0 + ldr) * (unsigned)K + (unsigned)ldo) * 2u;
  const unsigned boff = ((unsigned)(n0 + ldr) * (unsigned)K + (unsigned)ldo) * 2u;
  unsigned alds[2], blds[2];
  alds[0] = lds_addr_of(&as[0][ldr * GLDS2 + ldo]);
  alds[1] = lds_addr_of(&as[1][ldr * GLDS2 + ldo]);
  blds[0] = lds_addr_of(&bs[0][ldr * GLDS2 + ldo]);
  blds[1] = lds_addr_of(&bs[1][ldr * GLDS2 + ldo]);

  const v8f z8 = {0.f,0.f,0.f,0.f,0.f,0.f,0.f,0.f};
  v8f acc[2][4];
#pragma unroll
  for (int r = 0; r < 2; ++r)
#pragma unroll
    for (int c = 0; c < 4; ++c) acc[r][c] = z8;

  // preload tile 0
  async_copy64(alds[0], aoff, A);
  async_copy64(blds[0], boff, Bt);

  const int nt = K / 64;
  for (int kt = 0; kt < nt; ++kt) {
    wait_async0();      // my tile-kt transfers landed in LDS
    __syncthreads();    // everyone's landed; prev reads of other buffer done
    if (kt + 1 < nt) {  // prefetch next tile into the other buffer
      const unsigned d = (unsigned)(kt + 1) * 128u;  // 64 bf16 = 128 bytes
      async_copy64(alds[(kt + 1) & 1], aoff + d, A);
      async_copy64(blds[(kt + 1) & 1], boff + d, Bt);
    }
    const bf16* ab = as[kt & 1];
    const bf16* bb = bs[kt & 1];
#pragma unroll
    for (int ks = 0; ks < 2; ++ks) {
      v16bf af[2], bfr[4];
      af[0] = load_frag(ab + (wr      + fr) * GLDS2 + 32 * ks, hf);
      af[1] = load_frag(ab + (wr + 16 + fr) * GLDS2 + 32 * ks, hf);
#pragma unroll
      for (int c = 0; c < 4; ++c)
        bfr[c] = load_frag(bb + (wc + 16 * c + fr) * GLDS2 + 32 * ks, hf);
#pragma unroll
      for (int r = 0; r < 2; ++r)
#pragma unroll
        for (int c = 0; c < 4; ++c) acc[r][c] = wmma_bf16(af[r], bfr[c], acc[r][c]);
    }
  }

  // Epilogue. C/D layout: lane holds col = 16c + fr, vgpr v holds row v + 8*hf.
#pragma unroll
  for (int c = 0; c < 4; ++c) {
    const int col = n0 + wc + 16 * c + fr;
    const float bia = bias[col];
#pragma unroll
    for (int r = 0; r < 2; ++r) {
#pragma unroll
      for (int v = 0; v < 8; ++v) {
        const int row = m0 + wr + 16 * r + v + 8 * hf;
        float x = acc[r][c][v] + bia;
        if (do_gelu) x = 0.5f * x * (1.0f + erff(x * 0.70710678118654752f));
        const size_t idx = (size_t)row * N + col;
        if (resid) x += resid[idx];
        if (outF) outF[idx] = x;
        if (outB) outB[idx] = (bf16)x;
      }
    }
  }
}

// ---- flash attention: 128 queries / block, 64-key chunks, WMMA both GEMMs --
#define QLD 72  // 64 + 8 pad, rows are 144B (16B-aligned)

__global__ __launch_bounds__(256) void attention_k(const bf16* __restrict__ qkv,
                                                   bf16* __restrict__ O) {
  __shared__ bf16 qs[128 * QLD];  // Q block   [query][hd]
  __shared__ bf16 ks[64 * QLD];   // K chunk   [key][hd]
  __shared__ bf16 vs[64 * QLD];   // V chunk T [hd][key]
  __shared__ bf16 ps[128 * QLD];  // P         [8 waves][16 rows][64 keys]

  const int tid = threadIdx.x, lane = tid & 31, wv = tid >> 5;
  const int fr = lane & 15, hf = lane >> 4;
  const int bh = blockIdx.y, bb = bh / Hn, hh = bh % Hn;
  const size_t tok0 = (size_t)bb * Nseq;
  const int q0 = blockIdx.x * 128;
  const bf16* vb = qkv + tok0 * (3 * Cdim) + 2 * Cdim + hh * HD;

  {  // async load Q block: thread t -> row t/2, 32 bf16 (64B) half-row
    const int r = tid >> 1, off = (tid & 1) * 32;
    const unsigned goff =
        ((unsigned)(tok0 + q0 + r) * (3u * Cdim) + (unsigned)(hh * HD + off)) * 2u;
    async_copy64(lds_addr_of(&qs[r * QLD + off]), goff, qkv);
  }
  wait_async0();
  // wave w owns query rows 16w..16w+15: exactly what its lanes fetched.
  v16bf qf0 = load_frag(qs + (16 * wv + fr) * QLD, hf);       // hd 0..31
  v16bf qf1 = load_frag(qs + (16 * wv + fr) * QLD + 32, hf);  // hd 32..63

  const v8f z8 = {0.f,0.f,0.f,0.f,0.f,0.f,0.f,0.f};
  v8f ov[4];
#pragma unroll
  for (int c2 = 0; c2 < 4; ++c2) ov[c2] = z8;
  float mi[8], li[8];
#pragma unroll
  for (int r = 0; r < 8; ++r) { mi[r] = -1e30f; li[r] = 0.f; }
  bf16* myps = ps + wv * 16 * QLD;

  for (int j = 0; j < Nseq; j += 64) {
    __syncthreads();  // previous chunk's ks/vs reads complete
    {  // K chunk (async): thread t -> key t/4, 16 bf16 (32B) quarter-row
      const int r = tid >> 2, off = (tid & 3) * 16;
      const unsigned gk =
          ((unsigned)(tok0 + j + r) * (3u * Cdim) + (unsigned)(Cdim + hh * HD + off)) * 2u;
      async_copy32(lds_addr_of(&ks[r * QLD + off]), gk, qkv);
      // V chunk, transposed into [hd][key] (manual element scatter)
      const v8bf* gv = (const v8bf*)(vb + (size_t)(j + r) * (3 * Cdim) + off);
      v8bf v0 = gv[0], v1 = gv[1];
#pragma unroll
      for (int i = 0; i < 8; ++i) {
        vs[(off + i)     * QLD + r] = v0[i];
        vs[(off + 8 + i) * QLD + r] = v1[i];
      }
    }
    wait_async0();
    __syncthreads();

    // S = Q K^T  (16 rows x 64 keys per wave)
    v8f s[4];
#pragma unroll
    for (int c = 0; c < 4; ++c) {
      v16bf k0f = load_frag(ks + (16 * c + fr) * QLD, hf);
      v16bf k1f = load_frag(ks + (16 * c + fr) * QLD + 32, hf);
      v8f t = z8;
      t = wmma_bf16(qf0, k0f, t);
      t = wmma_bf16(qf1, k1f, t);
      s[c] = t;
    }

    // online softmax (rows live in vgpr index + 8*hf; reduce over 16 lanes)
    float mnew[8];
#pragma unroll
    for (int r = 0; r < 8; ++r) mnew[r] = mi[r];
#pragma unroll
    for (int c = 0; c < 4; ++c)
#pragma unroll
      for (int r = 0; r < 8; ++r) mnew[r] = fmaxf(mnew[r], s[c][r] * 0.125f);
#pragma unroll
    for (int r = 0; r < 8; ++r) {
      float t = mnew[r];
      t = fmaxf(t, __shfl_xor(t, 1));
      t = fmaxf(t, __shfl_xor(t, 2));
      t = fmaxf(t, __shfl_xor(t, 4));
      t = fmaxf(t, __shfl_xor(t, 8));
      mnew[r] = t;
    }
    float rsum[8];
#pragma unroll
    for (int r = 0; r < 8; ++r) {
      float a = __expf(mi[r] - mnew[r]);
      li[r] *= a; rsum[r] = 0.f; mi[r] = mnew[r];
#pragma unroll
      for (int c2 = 0; c2 < 4; ++c2) ov[c2][r] *= a;
    }
#pragma unroll
    for (int c = 0; c < 4; ++c)
#pragma unroll
      for (int r = 0; r < 8; ++r) {
        float pp = __expf(s[c][r] * 0.125f - mnew[r]);
        rsum[r] += pp;
        myps[(r + 8 * hf) * QLD + 16 * c + fr] = (bf16)pp;  // C->A layout via LDS
      }
#pragma unroll
    for (int r = 0; r < 8; ++r) {
      float t = rsum[r];
      t += __shfl_xor(t, 1); t += __shfl_xor(t, 2);
      t += __shfl_xor(t, 4); t += __shfl_xor(t, 8);
      li[r] += t;
    }

    // O += P V   (same-wave LDS write->read, in-order DS)
#pragma unroll
    for (int kk = 0; kk < 2; ++kk) {
      v16bf pf = load_frag(myps + fr * QLD + 32 * kk, hf);
#pragma unroll
      for (int c2 = 0; c2 < 4; ++c2) {
        v16bf vf = load_frag(vs + (16 * c2 + fr) * QLD + 32 * kk, hf);
        ov[c2] = wmma_bf16(pf, vf, ov[c2]);
      }
    }
  }

  // write O (bf16), normalized by row sums
#pragma unroll
  for (int c2 = 0; c2 < 4; ++c2)
#pragma unroll
    for (int r = 0; r < 8; ++r) {
      const size_t row = tok0 + q0 + 16 * wv + r + 8 * hf;
      const int ch = hh * HD + 16 * c2 + fr;
      O[row * Cdim + ch] = (bf16)(ov[c2][r] / li[r]);
    }
}

// ---- LayerNorm over C=768 (one block per token) ---------------------------
__global__ __launch_bounds__(256) void layernorm_k(const float* __restrict__ X,
                                                   const float* __restrict__ g,
                                                   const float* __restrict__ bta,
                                                   float* __restrict__ outF,
                                                   bf16* __restrict__ outB) {
  __shared__ float red[256];
  const int row = blockIdx.x, tid = threadIdx.x;
  const float* x = X + (size_t)row * Cdim;
  float v0 = x[tid], v1 = x[tid + 256], v2 = x[tid + 512];
  red[tid] = v0 + v1 + v2;
  __syncthreads();
  for (int o = 128; o > 0; o >>= 1) {
    if (tid < o) red[tid] += red[tid + o];
    __syncthreads();
  }
  const float mu = red[0] * (1.0f / Cdim);
  __syncthreads();
  float d0 = v0 - mu, d1 = v1 - mu, d2 = v2 - mu;
  red[tid] = d0 * d0 + d1 * d1 + d2 * d2;
  __syncthreads();
  for (int o = 128; o > 0; o >>= 1) {
    if (tid < o) red[tid] += red[tid + o];
    __syncthreads();
  }
  const float rstd = rsqrtf(red[0] * (1.0f / Cdim) + 1e-5f);
#pragma unroll
  for (int i = 0; i < 3; ++i) {
    const int c = tid + 256 * i;
    const float vv = (i == 0 ? v0 : (i == 1 ? v1 : v2));
    const float y = (vv - mu) * rstd * g[c] + bta[c];
    const size_t idx = (size_t)row * Cdim + c;
    if (outF) outF[idx] = y;
    if (outB) outB[idx] = (bf16)y;
  }
}

// ---------------------------------------------------------------------------
extern "C" void kernel_launch(void* const* d_in, const int* in_sizes, int n_in,
                              void* d_out, int out_size, void* d_ws, size_t ws_size,
                              hipStream_t stream) {
  (void)in_sizes; (void)n_in; (void)out_size; (void)ws_size;
  const float* x     = (const float*)d_in[0];
  const float* Wqkv  = (const float*)d_in[1];
  const float* bqkv  = (const float*)d_in[2];
  const float* Wproj = (const float*)d_in[3];
  const float* bproj = (const float*)d_in[4];
  const float* g1    = (const float*)d_in[5];
  const float* b1    = (const float*)d_in[6];
  const float* g2    = (const float*)d_in[7];
  const float* b2    = (const float*)d_in[8];
  const float* Wfc1  = (const float*)d_in[9];
  const float* bfc1  = (const float*)d_in[10];
  const float* Wfc2  = (const float*)d_in[11];
  const float* bfc2  = (const float*)d_in[12];
  float* out = (float*)d_out;

  // workspace partition (~216 MB total)
  char* p = (char*)d_ws;
  auto alloc = [&](size_t bytes) -> void* {
    void* r = (void*)p;
    p += (bytes + 255) & ~(size_t)255;
    return r;
  };
  bf16*  xb     = (bf16*)alloc((size_t)NTOK * Cdim * 2);
  bf16*  WqkvT  = (bf16*)alloc((size_t)3 * Cdim * Cdim * 2);
  bf16*  WprojT = (bf16*)alloc((size_t)Cdim * Cdim * 2);
  bf16*  Wfc1T  = (bf16*)alloc((size_t)DFF * Cdim * 2);
  bf16*  Wfc2T  = (bf16*)alloc((size_t)Cdim * DFF * 2);
  bf16*  qkvb   = (bf16*)alloc((size_t)NTOK * 3 * Cdim * 2);
  bf16*  Ob     = (bf16*)alloc((size_t)NTOK * Cdim * 2);
  float* y1     = (float*)alloc((size_t)NTOK * Cdim * 4);
  float* ln1f   = (float*)alloc((size_t)NTOK * Cdim * 4);
  bf16*  ln1b   = (bf16*)alloc((size_t)NTOK * Cdim * 2);
  bf16*  h1     = (bf16*)alloc((size_t)NTOK * DFF * 2);
  float* y2     = (float*)alloc((size_t)NTOK * Cdim * 4);

  const dim3 blk(256);

  {  // x -> bf16
    const int n = NTOK * Cdim;
    cvt_k<<<dim3((n / 4 + 255) / 256), blk, 0, stream>>>(x, xb, n);
  }
  // weights -> transposed bf16
  transpose_cvt_k<<<dim3(3 * Cdim / 32, Cdim / 32), blk, 0, stream>>>(Wqkv, WqkvT, Cdim, 3 * Cdim);
  transpose_cvt_k<<<dim3(Cdim / 32, Cdim / 32),     blk, 0, stream>>>(Wproj, WprojT, Cdim, Cdim);
  transpose_cvt_k<<<dim3(DFF / 32, Cdim / 32),      blk, 0, stream>>>(Wfc1, Wfc1T, Cdim, DFF);
  transpose_cvt_k<<<dim3(Cdim / 32, DFF / 32),      blk, 0, stream>>>(Wfc2, Wfc2T, DFF, Cdim);

  // qkv = x @ Wqkv + bqkv                          (bf16 out)
  gemm_bf16_k<<<dim3(3 * Cdim / 128, NTOK / 128), blk, 0, stream>>>(
      xb, WqkvT, bqkv, nullptr, nullptr, qkvb, NTOK, Cdim, 3 * Cdim, 0);
  // O = softmax(QK^T/8) V                          (bf16 out)
  attention_k<<<dim3(Nseq / 128, Bdim * Hn), blk, 0, stream>>>(qkvb, Ob);
  // y1 = x + O @ Wproj + bproj                     (f32 out)
  gemm_bf16_k<<<dim3(Cdim / 128, NTOK / 128), blk, 0, stream>>>(
      Ob, WprojT, bproj, x, y1, nullptr, NTOK, Cdim, Cdim, 0);
  // ln1 = LN(y1)                                   (f32 + bf16)
  layernorm_k<<<dim3(NTOK), blk, 0, stream>>>(y1, g1, b1, ln1f, ln1b);
  // h1 = gelu(ln1 @ Wfc1 + bfc1)                   (bf16 out)
  gemm_bf16_k<<<dim3(DFF / 128, NTOK / 128), blk, 0, stream>>>(
      ln1b, Wfc1T, bfc1, nullptr, nullptr, h1, NTOK, Cdim, DFF, 1);
  // y2 = ln1 + h1 @ Wfc2 + bfc2                    (f32 out)
  gemm_bf16_k<<<dim3(Cdim / 128, NTOK / 128), blk, 0, stream>>>(
      h1, Wfc2T, bfc2, ln1f, y2, nullptr, NTOK, DFF, Cdim, 0);
  // out = LN(y2)
  layernorm_k<<<dim3(NTOK), blk, 0, stream>>>(y2, g2, b2, out, nullptr);
}
